// TemporalDecoderBlock_22814866276432
// MI455X (gfx1250) — compile-verified, matrix operand order checked by reading, not run on hardware
//
#include <hip/hip_runtime.h>
#include <hip/hip_bf16.h>
#include <cstddef>

// ---------------- problem constants ----------------
#define B_    2
#define S_    2048
#define CTX_  2048
#define DIM_  2048
#define H_    16
#define HKV_  4
#define HD_   128
#define HID_  5464
#define EPS_  1.1920929e-07f
#define QKVO_ ((H_ + 2*HKV_) * HD_)   // 3072
#define KVO_  (2*HKV_*HD_)            // 1024
#define MS_   (B_ * S_)               // 4096 rows
#define MC_   (B_ * CTX_)             // 4096 rows
#define KPAD_ 5472                    // HID padded to mult of 32 (GEMM K)
#define NPAD_ 5504                    // HID padded to mult of 64 (GEMM N)

// ---------------- CDNA5 WMMA types ----------------
typedef __attribute__((ext_vector_type(16))) __bf16 v16bf;
typedef __attribute__((ext_vector_type(8)))  __bf16 v8bf;
typedef __attribute__((ext_vector_type(8)))  float  v8f;

__device__ __forceinline__ v8f vzero8() {
  v8f z = {0.f,0.f,0.f,0.f,0.f,0.f,0.f,0.f};
  return z;
}

__device__ __forceinline__ unsigned short f2bf(float f) {
  unsigned u = __float_as_uint(f);
  u += 0x7fffu + ((u >> 16) & 1u);          // round-to-nearest-even
  return (unsigned short)(u >> 16);
}

__device__ __forceinline__ v16bf join16(v8bf lo, v8bf hi) {
  return __builtin_shufflevector(lo, hi, 0,1,2,3,4,5,6,7,8,9,10,11,12,13,14,15);
}

// 16x32 bf16 A/B fragment slice for one lane from a row-major [rows,K] source:
// lane-half 0 takes K-chunks {0..7},{16..23}; half 1 takes {8..15},{24..31}.
__device__ __forceinline__ v16bf load_frag(const unsigned short* rowptr, int hi) {
  const unsigned short* p = rowptr + hi * 8;
  v8bf lo = *(const v8bf*)p;
  v8bf hh = *(const v8bf*)(p + 16);
  return join16(lo, hh);
}

__device__ __forceinline__ float wave_sum32(float v) {
#pragma unroll
  for (int o = 16; o > 0; o >>= 1) v += __shfl_xor(v, o, 32);
  return v;
}

// ---------------- elementwise cast / silu ----------------
__global__ void cast_bf16_kernel(const float* __restrict__ x,
                                 unsigned short* __restrict__ y, long long n) {
  long long i = (long long)blockIdx.x * blockDim.x + threadIdx.x;
  long long stride = (long long)gridDim.x * blockDim.x;
  for (; i < n; i += stride) y[i] = f2bf(x[i]);
}

// cast [rows,cols] fp32 -> [rowsPad,ldout] bf16, zero-filling the padding
__global__ void cast_pad_bf16_kernel(const float* __restrict__ x,
                                     unsigned short* __restrict__ y,
                                     int rows, int cols, int rowsPad, int ldout) {
  long long n = (long long)rowsPad * ldout;
  long long i = (long long)blockIdx.x * blockDim.x + threadIdx.x;
  long long stride = (long long)gridDim.x * blockDim.x;
  for (; i < n; i += stride) {
    int r = (int)(i / ldout), c = (int)(i - (long long)r * ldout);
    y[i] = (r < rows && c < cols) ? f2bf(x[(size_t)r * cols + c])
                                  : (unsigned short)0;
  }
}

// silu on g[MS_, HID_] stored with ld NPAD_, writing bf16 with ld KPAD_ (pad=0)
__global__ void silu_pad_kernel(const float* __restrict__ g,
                                unsigned short* __restrict__ y) {
  long long n = (long long)MS_ * KPAD_;
  long long i = (long long)blockIdx.x * blockDim.x + threadIdx.x;
  long long stride = (long long)gridDim.x * blockDim.x;
  for (; i < n; i += stride) {
    int r = (int)(i / KPAD_), c = (int)(i - (long long)r * KPAD_);
    if (c < HID_) {
      float v = g[(size_t)r * NPAD_ + c];
      y[i] = f2bf(v / (1.f + __expf(-v)));
    } else {
      y[i] = 0;
    }
  }
}

// ---------------- RMSNorm (row length DIM_) + bf16 cast ----------------
__global__ __launch_bounds__(256)
void rmsnorm_cast_kernel(const float* __restrict__ x, const float* __restrict__ w,
                         unsigned short* __restrict__ y) {
  __shared__ float red[8];
  const int row = blockIdx.x;
  const int tid = threadIdx.x;
  const float* xr = x + (size_t)row * DIM_;
  unsigned short* yr = y + (size_t)row * DIM_;
  float vals[8];
  float ss = 0.f;
#pragma unroll
  for (int e = 0; e < 8; ++e) {
    float v = xr[tid * 8 + e];
    vals[e] = v;
    ss += v * v;
  }
  ss = wave_sum32(ss);
  if ((tid & 31) == 0) red[tid >> 5] = ss;
  __syncthreads();
  float tot = 0.f;
#pragma unroll
  for (int i = 0; i < 8; ++i) tot += red[i];
  float r = rsqrtf(tot * (1.f / DIM_) + EPS_);
#pragma unroll
  for (int e = 0; e < 8; ++e)
    yr[tid * 8 + e] = f2bf(vals[e] * r * w[tid * 8 + e]);
}

// ---------------- WMMA bf16 GEMM:  C[M,N] = A[M,K] * B[N,K]^T (+res) -------------
// 256 threads = 8 waves, 128x64 tile, K-tiles of 32, double-buffered LDS:
// iteration kt stages tile kt+1 into the other buffer while WMMAs consume tile kt
// (one barrier per K-step).  PRECONDITIONS (guaranteed by padded workspace):
//   gridDim.y*128 rows / gridDim.x*64 cols fully allocated, K % 32 == 0,
//   all row strides 16B-aligned.  No guards anywhere.
__global__ __launch_bounds__(256)
void gemm_bf16_nt(const unsigned short* __restrict__ A,
                  const unsigned short* __restrict__ Bm,
                  const float* __restrict__ res,
                  float* __restrict__ C,
                  int K, int ldc) {
  __shared__ __align__(16) unsigned short As[2][128][48];  // row stride 96B
  __shared__ __align__(16) unsigned short Bs[2][64][48];
  const int tid  = threadIdx.x;
  const int wave = tid >> 5, lane = tid & 31;
  const int hi = lane >> 4, lr = lane & 15;
  const int wm = wave >> 1, wn = wave & 1;      // 4 x 2 wave quadrants of 32x32
  const size_t tileM = (size_t)blockIdx.y * 128;
  const size_t tileN = (size_t)blockIdx.x * 64;

  // A staging: 128 rows x 32 cols, 16 bf16 per thread
  const int arow = tid >> 1;
  const int acol = (tid & 1) << 4;
  // B staging: 64 rows x 32 cols, 8 bf16 per thread
  const int brow = tid >> 2;
  const int bcol = (tid & 3) << 3;

  const unsigned short* aptr = A + (tileM + arow) * (size_t)K + acol;
  const unsigned short* bptr = Bm + (tileN + brow) * (size_t)K + bcol;

  v8f acc[2][2];
#pragma unroll
  for (int i = 0; i < 2; ++i)
#pragma unroll
    for (int j = 0; j < 2; ++j) acc[i][j] = vzero8();

  // stage tile 0 into buffer 0
  *(v8bf*)&As[0][arow][acol]     = *(const v8bf*)aptr;
  *(v8bf*)&As[0][arow][acol + 8] = *(const v8bf*)(aptr + 8);
  *(v8bf*)&Bs[0][brow][bcol]     = *(const v8bf*)bptr;
  aptr += 32;
  bptr += 32;
  __syncthreads();

  const int nk = K >> 5;
  for (int kt = 0; kt < nk; ++kt) {
    const int cur = kt & 1;
    if (kt + 1 < nk) {
      // distance-2 prefetch into near caches (gfx1250 global_prefetch_b8)
      __builtin_prefetch(aptr + 32, 0, 3);
      __builtin_prefetch(bptr + 32, 0, 3);
      // stage tile kt+1 into the other buffer while this tile computes
      *(v8bf*)&As[cur ^ 1][arow][acol]     = *(const v8bf*)aptr;
      *(v8bf*)&As[cur ^ 1][arow][acol + 8] = *(const v8bf*)(aptr + 8);
      *(v8bf*)&Bs[cur ^ 1][brow][bcol]     = *(const v8bf*)bptr;
      aptr += 32;
      bptr += 32;
    }

    v16bf afrag[2], bfrag[2];
#pragma unroll
    for (int i = 0; i < 2; ++i)
      afrag[i] = load_frag(&As[cur][wm * 32 + i * 16 + lr][0], hi);
#pragma unroll
    for (int j = 0; j < 2; ++j)
      bfrag[j] = load_frag(&Bs[cur][wn * 32 + j * 16 + lr][0], hi);

#pragma unroll
    for (int i = 0; i < 2; ++i)
#pragma unroll
      for (int j = 0; j < 2; ++j)
        acc[i][j] = __builtin_amdgcn_wmma_f32_16x16x32_bf16(
            false, afrag[i], false, bfrag[j], (short)0, acc[i][j], false, false);
    __syncthreads();
  }

  // epilogue: C layout -> row = base + v + 8*hi, col = base + lr  (unguarded)
#pragma unroll
  for (int i = 0; i < 2; ++i) {
#pragma unroll
    for (int j = 0; j < 2; ++j) {
      size_t colg = tileN + wn * 32 + j * 16 + lr;
#pragma unroll
      for (int v = 0; v < 8; ++v) {
        size_t rowg = tileM + wm * 32 + i * 16 + v + hi * 8;
        float val = acc[i][j][v];
        if (res) val += res[rowg * ldc + colg];
        C[rowg * ldc + colg] = val;
      }
    }
  }
}

// ---------------- RoPE / QK-norm producers ----------------
// Self path: norm -> rope.  Q gets 1/sqrt(HD) folded in.  V stored transposed.
__global__ __launch_bounds__(128)
void rope_self_kernel(const float* __restrict__ qkv,
                      const float* __restrict__ qn, const float* __restrict__ kn,
                      const float* __restrict__ fcos, const float* __restrict__ fsin,
                      unsigned short* __restrict__ qo,
                      unsigned short* __restrict__ ko,
                      unsigned short* __restrict__ vTo) {
  const int s = blockIdx.x, b = blockIdx.y;
  const int wave = threadIdx.x >> 5, lane = threadIdx.x & 31;
  const int d0 = lane * 4;
  const float* base = qkv + ((size_t)b * S_ + s) * QKVO_;
  float c[4], sn[4];
#pragma unroll
  for (int e = 0; e < 4; ++e) {
    c[e]  = fcos[(size_t)s * HD_ + d0 + e];
    sn[e] = fsin[(size_t)s * HD_ + d0 + e];
  }
  const float qs = 0.08838834764831845f;  // 1/sqrt(128)

  for (int hh = wave; hh < H_; hh += 4) {
    const float* p = base + hh * HD_ + d0;
    float v0 = p[0], v1 = p[1], v2 = p[2], v3 = p[3];
    float r = rsqrtf(wave_sum32(v0*v0 + v1*v1 + v2*v2 + v3*v3) * (1.f/HD_) + EPS_);
    float y0 = v0*r*qn[d0+0], y1 = v1*r*qn[d0+1], y2 = v2*r*qn[d0+2], y3 = v3*r*qn[d0+3];
    unsigned short* qp = qo + (((size_t)b * H_ + hh) * S_ + s) * HD_ + d0;
    qp[0] = f2bf((y0*c[0] - y1*sn[0]) * qs);
    qp[1] = f2bf((y1*c[1] + y0*sn[1]) * qs);
    qp[2] = f2bf((y2*c[2] - y3*sn[2]) * qs);
    qp[3] = f2bf((y3*c[3] + y2*sn[3]) * qs);
  }
  {
    const int kh = wave;
    const float* p = base + (H_ + kh) * HD_ + d0;
    float v0 = p[0], v1 = p[1], v2 = p[2], v3 = p[3];
    float r = rsqrtf(wave_sum32(v0*v0 + v1*v1 + v2*v2 + v3*v3) * (1.f/HD_) + EPS_);
    float y0 = v0*r*kn[d0+0], y1 = v1*r*kn[d0+1], y2 = v2*r*kn[d0+2], y3 = v3*r*kn[d0+3];
    unsigned short* kp = ko + (((size_t)b * HKV_ + kh) * S_ + s) * HD_ + d0;
    kp[0] = f2bf(y0*c[0] - y1*sn[0]);
    kp[1] = f2bf(y1*c[1] + y0*sn[1]);
    kp[2] = f2bf(y2*c[2] - y3*sn[2]);
    kp[3] = f2bf(y3*c[3] + y2*sn[3]);
  }
  {
    const int vh = wave;
    const float* p = base + (H_ + HKV_ + vh) * HD_ + d0;
#pragma unroll
    for (int e = 0; e < 4; ++e)
      vTo[(((size_t)b * HKV_ + vh) * HD_ + d0 + e) * (size_t)S_ + s] = f2bf(p[e]);
  }
}

// Cross-Q path: rope FIRST, then rmsnorm, then scale.
__global__ __launch_bounds__(128)
void rope_cross_q_kernel(const float* __restrict__ xq,
                         const float* __restrict__ qn,
                         const float* __restrict__ fcos, const float* __restrict__ fsin,
                         unsigned short* __restrict__ qo) {
  const int s = blockIdx.x, b = blockIdx.y;
  const int wave = threadIdx.x >> 5, lane = threadIdx.x & 31;
  const int d0 = lane * 4;
  const float* base = xq + ((size_t)b * S_ + s) * (H_ * HD_);
  float c[4], sn[4];
#pragma unroll
  for (int e = 0; e < 4; ++e) {
    c[e]  = fcos[(size_t)s * HD_ + d0 + e];
    sn[e] = fsin[(size_t)s * HD_ + d0 + e];
  }
  const float qs = 0.08838834764831845f;
  for (int hh = wave; hh < H_; hh += 4) {
    const float* p = base + hh * HD_ + d0;
    float v0 = p[0], v1 = p[1], v2 = p[2], v3 = p[3];
    float o0 = v0*c[0] - v1*sn[0];
    float o1 = v1*c[1] + v0*sn[1];
    float o2 = v2*c[2] - v3*sn[2];
    float o3 = v3*c[3] + v2*sn[3];
    float r = rsqrtf(wave_sum32(o0*o0 + o1*o1 + o2*o2 + o3*o3) * (1.f/HD_) + EPS_);
    unsigned short* qp = qo + (((size_t)b * H_ + hh) * S_ + s) * HD_ + d0;
    qp[0] = f2bf(o0 * r * qn[d0+0] * qs);
    qp[1] = f2bf(o1 * r * qn[d0+1] * qs);
    qp[2] = f2bf(o2 * r * qn[d0+2] * qs);
    qp[3] = f2bf(o3 * r * qn[d0+3] * qs);
  }
}

// Cross KV: k gets rmsnorm only (no rope), v stored transposed.
__global__ __launch_bounds__(128)
void cross_kv_kernel(const float* __restrict__ kv,
                     const float* __restrict__ kn,
                     unsigned short* __restrict__ ko,
                     unsigned short* __restrict__ vTo) {
  const int t = blockIdx.x, b = blockIdx.y;
  const int wave = threadIdx.x >> 5, lane = threadIdx.x & 31;
  const int d0 = lane * 4;
  const float* base = kv + ((size_t)b * CTX_ + t) * KVO_;
  {
    const float* p = base + wave * HD_ + d0;
    float v0 = p[0], v1 = p[1], v2 = p[2], v3 = p[3];
    float r = rsqrtf(wave_sum32(v0*v0 + v1*v1 + v2*v2 + v3*v3) * (1.f/HD_) + EPS_);
    unsigned short* kp = ko + (((size_t)b * HKV_ + wave) * CTX_ + t) * HD_ + d0;
    kp[0] = f2bf(v0 * r * kn[d0+0]);
    kp[1] = f2bf(v1 * r * kn[d0+1]);
    kp[2] = f2bf(v2 * r * kn[d0+2]);
    kp[3] = f2bf(v3 * r * kn[d0+3]);
  }
  {
    const float* p = base + (HKV_ + wave) * HD_ + d0;
#pragma unroll
    for (int e = 0; e < 4; ++e)
      vTo[(((size_t)b * HKV_ + wave) * HD_ + d0 + e) * (size_t)CTX_ + t] = f2bf(p[e]);
  }
}

// ---------------- Flash attention (wave32, WMMA bf16) ----------------
// q:[B,H,S,HD] (scale folded), k:[B,HKV,L,HD], vT:[B,HKV,HD,L], out:[B,S,H,HD] f32
__global__ __launch_bounds__(32)
void flash_attn_bf16(const unsigned short* __restrict__ q,
                     const unsigned short* __restrict__ k,
                     const unsigned short* __restrict__ vT,
                     float* __restrict__ out,
                     const unsigned char* __restrict__ qmask,
                     const unsigned char* __restrict__ kmask,
                     int L) {
  __shared__ __align__(16) unsigned short P[16][32];  // 16x32 bf16 probabilities
  const int lane = threadIdx.x & 31;
  const int hi = lane >> 4, lr = lane & 15;
  const int q0 = blockIdx.x * 16;
  const int h = blockIdx.y, b = blockIdx.z;
  const int kvh = h / (H_ / HKV_);

  const unsigned short* qbase = q + (((size_t)b * H_ + h) * S_ + q0 + lr) * HD_;
  v16bf qf[4];
#pragma unroll
  for (int c = 0; c < 4; ++c) qf[c] = load_frag(qbase + c * 32, hi);

  v8f o[8];
#pragma unroll
  for (int n = 0; n < 8; ++n) o[n] = vzero8();
  float m_[8], l_[8];
#pragma unroll
  for (int v = 0; v < 8; ++v) { m_[v] = -3.0e38f; l_[v] = 0.f; }

  const unsigned short* kbase = k  + ((size_t)b * HKV_ + kvh) * (size_t)L * HD_;
  const unsigned short* vbase = vT + ((size_t)b * HKV_ + kvh) * (size_t)HD_ * L;
  const unsigned char* km = kmask ? (kmask + (size_t)b * L) : (const unsigned char*)0;

  for (int j = 0; j < L; j += 32) {
    v8f s0 = vzero8(), s1 = vzero8();
#pragma unroll
    for (int c = 0; c < 4; ++c) {
      v16bf kf = load_frag(kbase + (size_t)(j + lr) * HD_ + c * 32, hi);
      s0 = __builtin_amdgcn_wmma_f32_16x16x32_bf16(false, qf[c], false, kf,
                                                   (short)0, s0, false, false);
    }
#pragma unroll
    for (int c = 0; c < 4; ++c) {
      v16bf kf = load_frag(kbase + (size_t)(j + 16 + lr) * HD_ + c * 32, hi);
      s1 = __builtin_amdgcn_wmma_f32_16x16x32_bf16(false, qf[c], false, kf,
                                                   (short)0, s1, false, false);
    }
    float bias0 = 0.f, bias1 = 0.f;
    if (km) {
      if (!km[j + lr])      bias0 = -1e30f;
      if (!km[j + 16 + lr]) bias1 = -1e30f;
    }
    float alpha[8];
#pragma unroll
    for (int v = 0; v < 8; ++v) {
      float a0 = s0[v] + bias0;
      float a1 = s1[v] + bias1;
      float mx = fmaxf(a0, a1);
#pragma unroll
      for (int off = 8; off > 0; off >>= 1) mx = fmaxf(mx, __shfl_xor(mx, off, 32));
      float mn = fmaxf(m_[v], mx);
      alpha[v] = __expf(m_[v] - mn);
      float p0 = __expf(a0 - mn);
      float p1 = __expf(a1 - mn);
      float sum = p0 + p1;
#pragma unroll
      for (int off = 8; off > 0; off >>= 1) sum += __shfl_xor(sum, off, 32);
      l_[v] = l_[v] * alpha[v] + sum;
      m_[v] = mn;
      P[v + hi * 8][lr]      = f2bf(p0);
      P[v + hi * 8][lr + 16] = f2bf(p1);
    }
    __syncthreads();
    v16bf pf = load_frag(&P[lr][0], hi);     // re-layout C -> A fragment
#pragma unroll
    for (int n = 0; n < 8; ++n) {
#pragma unroll
      for (int v = 0; v < 8; ++v) o[n][v] *= alpha[v];
      v16bf vf = load_frag(vbase + (size_t)(n * 16 + lr) * L + j, hi);
      o[n] = __builtin_amdgcn_wmma_f32_16x16x32_bf16(false, pf, false, vf,
                                                     (short)0, o[n], false, false);
    }
    __syncthreads();
  }

#pragma unroll
  for (int v = 0; v < 8; ++v) {
    int row = q0 + v + hi * 8;
    float keep = 1.f;
    if (qmask && !qmask[(size_t)b * S_ + row]) keep = 0.f;
    float inv = keep / l_[v];
#pragma unroll
    for (int n = 0; n < 8; ++n)
      out[(((size_t)b * S_ + row) * H_ + h) * HD_ + n * 16 + lr] = o[n][v] * inv;
  }
}

// ---------------- host orchestration ----------------
extern "C" void kernel_launch(void* const* d_in, const int* in_sizes, int n_in,
                              void* d_out, int out_size, void* d_ws, size_t ws_size,
                              hipStream_t stream) {
  (void)in_sizes; (void)n_in; (void)out_size; (void)ws_size;
  const float* x            = (const float*)d_in[0];
  const float* context      = (const float*)d_in[1];
  const float* fcos         = (const float*)d_in[2];
  const float* fsin         = (const float*)d_in[3];
  const float* wqkv         = (const float*)d_in[4];
  const float* wo_self      = (const float*)d_in[5];
  const float* qn_self      = (const float*)d_in[6];
  const float* kn_self      = (const float*)d_in[7];
  const float* wq           = (const float*)d_in[8];
  const float* wkv          = (const float*)d_in[9];
  const float* wo_cross     = (const float*)d_in[10];
  const float* qn_cross     = (const float*)d_in[11];
  const float* kn_cross     = (const float*)d_in[12];
  const float* w1           = (const float*)d_in[13];
  const float* w2           = (const float*)d_in[14];
  const float* attn_norm_w  = (const float*)d_in[15];
  const float* cross_norm_w = (const float*)d_in[16];
  const float* ffn_norm_w   = (const float*)d_in[17];
  const unsigned char* vis_mask     = (const unsigned char*)d_in[18];
  const unsigned char* context_mask = (const unsigned char*)d_in[19];
  float* outp = (float*)d_out;

  // ---- workspace layout (~335 MB) ----
  char* ws = (char*)d_ws;
  size_t off = 0;
  auto alloc = [&](size_t bytes) -> void* {
    void* p = ws + off;
    off += (bytes + 255) & ~(size_t)255;
    return p;
  };
  unsigned short* wqkv_bf = (unsigned short*)alloc((size_t)QKVO_ * DIM_ * 2);
  unsigned short* wos_bf  = (unsigned short*)alloc((size_t)DIM_ * DIM_ * 2);
  unsigned short* wq_bf   = (unsigned short*)alloc((size_t)DIM_ * DIM_ * 2);
  unsigned short* wkv_bf  = (unsigned short*)alloc((size_t)KVO_ * DIM_ * 2);
  unsigned short* woc_bf  = (unsigned short*)alloc((size_t)DIM_ * DIM_ * 2);
  unsigned short* w1_bf   = (unsigned short*)alloc((size_t)NPAD_ * DIM_ * 2);  // rows padded
  unsigned short* w2_bf   = (unsigned short*)alloc((size_t)DIM_ * KPAD_ * 2);  // cols padded
  unsigned short* act_bf  = (unsigned short*)alloc((size_t)MS_ * DIM_ * 2);    // h0/attn/hc/ctx/hf
  float*          big_f   = (float*)alloc((size_t)MS_ * NPAD_ * 4);            // qkv | qc+kvc | g
  unsigned short* q_bf    = (unsigned short*)alloc((size_t)B_ * H_ * S_ * HD_ * 2);
  unsigned short* k_bf    = (unsigned short*)alloc((size_t)B_ * HKV_ * S_ * HD_ * 2);
  unsigned short* vT_bf   = (unsigned short*)alloc((size_t)B_ * HKV_ * HD_ * S_ * 2);
  float*          attn_f  = (float*)alloc((size_t)MS_ * DIM_ * 4);
  float*          h_f     = (float*)alloc((size_t)MS_ * DIM_ * 4);
  unsigned short* gs_bf   = (unsigned short*)alloc((size_t)MS_ * KPAD_ * 2);

  float* qkv_f = big_f;                       // [MS, 3072]
  float* qc_f  = big_f;                       // [MS, 2048]
  float* kvc_f = big_f + (size_t)MS_ * DIM_;  // [MC, 1024]
  float* g_f   = big_f;                       // [MS, HID] with ld NPAD_

  auto ggrid = [](int M, int Npad) { return dim3((unsigned)(Npad / 64),
                                                 (unsigned)(M / 128)); };
  const dim3 fgrid(S_ / 16, H_, B_);

  // ---- weight casts fp32 -> bf16 (padded where needed) ----
  cast_bf16_kernel<<<2048, 256, 0, stream>>>(wqkv, wqkv_bf, (long long)QKVO_ * DIM_);
  cast_bf16_kernel<<<2048, 256, 0, stream>>>(wo_self, wos_bf, (long long)DIM_ * DIM_);
  cast_bf16_kernel<<<2048, 256, 0, stream>>>(wq, wq_bf, (long long)DIM_ * DIM_);
  cast_bf16_kernel<<<2048, 256, 0, stream>>>(wkv, wkv_bf, (long long)KVO_ * DIM_);
  cast_bf16_kernel<<<2048, 256, 0, stream>>>(wo_cross, woc_bf, (long long)DIM_ * DIM_);
  cast_pad_bf16_kernel<<<2048, 256, 0, stream>>>(w1, w1_bf, HID_, DIM_, NPAD_, DIM_);
  cast_pad_bf16_kernel<<<2048, 256, 0, stream>>>(w2, w2_bf, DIM_, HID_, DIM_, KPAD_);

  // ---- self attention ----
  rmsnorm_cast_kernel<<<MS_, 256, 0, stream>>>(x, attn_norm_w, act_bf);          // h0
  gemm_bf16_nt<<<ggrid(MS_, QKVO_), 256, 0, stream>>>(act_bf, wqkv_bf, nullptr,
                                                      qkv_f, DIM_, QKVO_);       // qkv
  rope_self_kernel<<<dim3(S_, B_), 128, 0, stream>>>(qkv_f, qn_self, kn_self,
                                                     fcos, fsin, q_bf, k_bf, vT_bf);
  flash_attn_bf16<<<fgrid, 32, 0, stream>>>(q_bf, k_bf, vT_bf, attn_f,
                                            vis_mask, vis_mask, S_);
  cast_bf16_kernel<<<2048, 256, 0, stream>>>(attn_f, act_bf, (long long)MS_ * DIM_);
  gemm_bf16_nt<<<ggrid(MS_, DIM_), 256, 0, stream>>>(act_bf, wos_bf, x,
                                                     h_f, DIM_, DIM_);           // h = x + a@woT

  // ---- cross attention ----
  rmsnorm_cast_kernel<<<MS_, 256, 0, stream>>>(h_f, cross_norm_w, act_bf);       // hc
  gemm_bf16_nt<<<ggrid(MS_, DIM_), 256, 0, stream>>>(act_bf, wq_bf, nullptr,
                                                     qc_f, DIM_, DIM_);          // qc
  cast_bf16_kernel<<<2048, 256, 0, stream>>>(context, act_bf, (long long)MC_ * DIM_);
  gemm_bf16_nt<<<ggrid(MC_, KVO_), 256, 0, stream>>>(act_bf, wkv_bf, nullptr,
                                                     kvc_f, DIM_, KVO_);         // kv
  rope_cross_q_kernel<<<dim3(S_, B_), 128, 0, stream>>>(qc_f, qn_cross,
                                                        fcos, fsin, q_bf);
  cross_kv_kernel<<<dim3(CTX_, B_), 128, 0, stream>>>(kvc_f, kn_cross, k_bf, vT_bf);
  flash_attn_bf16<<<fgrid, 32, 0, stream>>>(q_bf, k_bf, vT_bf, attn_f,
                                            nullptr, context_mask, CTX_);
  cast_bf16_kernel<<<2048, 256, 0, stream>>>(attn_f, act_bf, (long long)MS_ * DIM_);
  gemm_bf16_nt<<<ggrid(MS_, DIM_), 256, 0, stream>>>(act_bf, woc_bf, h_f,
                                                     h_f, DIM_, DIM_);           // h += a@wocT

  // ---- FFN (silu MLP), padded N/K so the GEMM stays guard-free ----
  rmsnorm_cast_kernel<<<MS_, 256, 0, stream>>>(h_f, ffn_norm_w, act_bf);         // hf
  gemm_bf16_nt<<<ggrid(MS_, NPAD_), 256, 0, stream>>>(act_bf, w1_bf, nullptr,
                                                      g_f, DIM_, NPAD_);         // g (ld NPAD_)
  silu_pad_kernel<<<2048, 256, 0, stream>>>(g_f, gs_bf);                         // ld KPAD_
  gemm_bf16_nt<<<ggrid(MS_, DIM_), 256, 0, stream>>>(gs_bf, w2_bf, h_f,
                                                     outp, KPAD_, DIM_);         // out
}
